// BehaviorSpecificPFF_54219667145019
// MI455X (gfx1250) — matrix operand order repaired and checked by reading, not compiled, not run on hardware
//
#include <hip/hip_runtime.h>
#include <hip/hip_bf16.h>

typedef __attribute__((ext_vector_type(16))) __bf16 v16bf;
typedef __attribute__((ext_vector_type(8)))  float  v8f;

#define DM   256
#define DF   1024
#define NB   4
#define NTOK (32 * 2048)
#define TM   64
#define KC   64
#define NCHUNK (DF / KC)        // 16
#define SDM  (DM + 8)           // padded stride for K=256 images
#define SKC  (KC + 8)           // padded stride for K=64 images

#define W1_IMG_ELEMS (KC * SDM) // 16896 bf16 per (expert, chunk)
#define W2_IMG_ELEMS (DM * SKC) // 18432 bf16 per (expert, chunk)

// workspace offsets (bytes, all 16B aligned)
#define WS_COUNTS 0
#define WS_LISTS  256
#define WS_W1T    (WS_LISTS + NB * NTOK * 4)                      // 1048832
#define WS_W2T    (WS_W1T + NB * NCHUNK * W1_IMG_ELEMS * 2)       // +2162688

// ---------------- WMMA operand loaders (bf16, 16x16x32) ----------------
// A (16x32, MxK): lane m=lane&15; lanes<16 hold K {0..7, 16..23}, lanes>=16 hold {8..15, 24..31}
__device__ __forceinline__ v16bf load_a16x32(const __bf16* base, int row0, int stride,
                                             int kOff, int lane) {
  const int m  = lane & 15;
  const int kb = (lane >> 4) << 3;            // 0 or 8
  const __bf16* p = base + (row0 + m) * stride + kOff + kb;
  union { v16bf v; uint4 u[2]; } r;
  r.u[0] = *(const uint4*)(p);                // K = kb .. kb+7
  r.u[1] = *(const uint4*)(p + 16);           // K = kb+16 .. kb+23
  return r.v;
}

// B (32x16, KxN) from a transposed [N][K] image:
// lane n=lane&15; lanes<16 hold K {0..7, 16..23}, lanes>=16 hold {8..15, 24..31}
__device__ __forceinline__ v16bf load_b32x16(const __bf16* base, int col0, int stride,
                                             int kOff, int lane) {
  const int n  = lane & 15;
  const int kb = (lane >> 4) << 3;            // 0 or 8
  const __bf16* p = base + (col0 + n) * stride + kOff + kb;
  union { v16bf v; uint4 u[2]; } r;
  r.u[0] = *(const uint4*)(p);
  r.u[1] = *(const uint4*)(p + 16);
  return r.v;
}

__device__ __forceinline__ void store_bf8(__bf16* dst, const float4& a, const float4& b) {
  union { __bf16 h[8]; uint4 u; } t;
  t.h[0] = (__bf16)a.x; t.h[1] = (__bf16)a.y; t.h[2] = (__bf16)a.z; t.h[3] = (__bf16)a.w;
  t.h[4] = (__bf16)b.x; t.h[5] = (__bf16)b.y; t.h[6] = (__bf16)b.z; t.h[7] = (__bf16)b.w;
  *(uint4*)dst = t.u;
}

// ---------------- Pass 0: zero output + counters ----------------
__global__ void bpff_zero_kernel(float4* __restrict__ out4, int n4, int* __restrict__ counts) {
  int i = blockIdx.x * blockDim.x + threadIdx.x;
  if (i < n4) out4[i] = make_float4(0.f, 0.f, 0.f, 0.f);
  if (blockIdx.x == 0 && threadIdx.x < NB) counts[threadIdx.x] = 0;
}

// ---------------- Pass 1: per-expert token index lists ----------------
__global__ void bpff_build_lists(const int* __restrict__ b_seq, int* __restrict__ counts,
                                 int* __restrict__ lists) {
  int t = blockIdx.x * blockDim.x + threadIdx.x;
  if (t < NTOK) {
    int b = b_seq[t];
    if (b > 0) {
      int idx = atomicAdd(&counts[b - 1], 1);
      lists[(b - 1) * NTOK + idx] = t;
    }
  }
}

// ---------------- Pass 1.5: pack weights once -> bf16 transposed LDS images ----------------
// One block per (chunk, expert). Transposes through LDS so global reads stay coalesced.
__global__ __launch_bounds__(256)
void bpff_pack_weights(const float* __restrict__ W1, const float* __restrict__ W2,
                       __bf16* __restrict__ w1t, __bf16* __restrict__ w2t) {
  __shared__ __bf16 sbuf[W2_IMG_ELEMS];       // 36864 B, >= W1_IMG_ELEMS too
  const int c   = blockIdx.x;                 // chunk
  const int e   = blockIdx.y;                 // expert
  const int fc  = c * KC;
  const int tid = threadIdx.x;

  // zero pads for deterministic workspace contents
  {
    uint4* z = (uint4*)sbuf;
    #pragma unroll
    for (int idx = tid; idx < W2_IMG_ELEMS / 8; idx += 256) z[idx] = make_uint4(0, 0, 0, 0);
  }
  __syncthreads();

  // ---- W1^T image: sbuf[f_local * SDM + i] = bf16(W1[e][i][fc+f_local]) ----
  #pragma unroll
  for (int j = 0; j < 16; ++j) {
    int idx = tid + 256 * j;
    int i   = idx >> 4;
    int fl4 = (idx & 15) * 4;
    float4 v = *(const float4*)(W1 + ((size_t)e * DM + i) * DF + fc + fl4);
    sbuf[(fl4 + 0) * SDM + i] = (__bf16)v.x;
    sbuf[(fl4 + 1) * SDM + i] = (__bf16)v.y;
    sbuf[(fl4 + 2) * SDM + i] = (__bf16)v.z;
    sbuf[(fl4 + 3) * SDM + i] = (__bf16)v.w;
  }
  __syncthreads();
  {
    const uint4* s = (const uint4*)sbuf;
    uint4* d = (uint4*)(w1t + (size_t)(e * NCHUNK + c) * W1_IMG_ELEMS);
    for (int idx = tid; idx < W1_IMG_ELEMS / 8; idx += 256) d[idx] = s[idx];
  }
  __syncthreads();

  // ---- W2^T image: sbuf[o * SKC + f_local] = bf16(W2[e][fc+f_local][o]) ----
  #pragma unroll
  for (int j = 0; j < 16; ++j) {
    int idx = tid + 256 * j;
    int fl  = idx >> 6;
    int o4  = (idx & 63) * 4;
    float4 v = *(const float4*)(W2 + ((size_t)e * DF + fc + fl) * DM + o4);
    sbuf[(o4 + 0) * SKC + fl] = (__bf16)v.x;
    sbuf[(o4 + 1) * SKC + fl] = (__bf16)v.y;
    sbuf[(o4 + 2) * SKC + fl] = (__bf16)v.z;
    sbuf[(o4 + 3) * SKC + fl] = (__bf16)v.w;
  }
  __syncthreads();
  {
    const uint4* s = (const uint4*)sbuf;
    uint4* d = (uint4*)(w2t + (size_t)(e * NCHUNK + c) * W2_IMG_ELEMS);
    #pragma unroll
    for (int idx = tid; idx < W2_IMG_ELEMS / 8; idx += 256) d[idx] = s[idx];
  }
}

// ---------------- Pass 2: gathered grouped-GEMM FFN ----------------
__global__ __launch_bounds__(256)
void bpff_ffn_kernel(const float* __restrict__ x,
                     const __bf16* __restrict__ w1t, const float* __restrict__ b1,
                     const __bf16* __restrict__ w2t, const float* __restrict__ b2,
                     const int* __restrict__ counts, const int* __restrict__ lists,
                     float* __restrict__ out) {
  extern __shared__ char smem[];
  int*    sTid = (int*)smem;                              // 64 ints   (256 B)
  float*  sB1  = (float*)(smem + 256);                    // 64 floats (256 B)
  __bf16* sX   = (__bf16*)(smem + 512);                   // [TM][SDM]
  __bf16* sW1  = sX  + TM * SDM;                          // [KC][SDM]  (W1^T chunk)
  __bf16* sW2  = sW1 + KC * SDM;                          // [DM][SKC]  (W2^T chunk)
  __bf16* sH   = sW2 + DM * SKC;                          // [TM][SKC]

  const int e        = blockIdx.y;
  const int cnt      = counts[e];
  const int tileBase = blockIdx.x * TM;
  if (tileBase >= cnt) return;

  const int tid  = threadIdx.x;
  const int lane = tid & 31;
  const int wave = tid >> 5;

  // token ids for this tile (tail rows alias token 0; masked at store)
  if (tid < TM) {
    int idx = tileBase + tid;
    sTid[tid] = (idx < cnt) ? lists[e * NTOK + idx] : 0;
  }
  __syncthreads();

  // stage gathered X rows -> bf16 LDS (4 threads per row, 64 cols each)
  {
    const int r  = tid >> 2;
    const int c0 = (tid & 3) * 64;
    const float* src = x + (size_t)sTid[r] * DM + c0;
    __bf16* dst = sX + r * SDM + c0;
    #pragma unroll
    for (int c = 0; c < 64; c += 8) {
      float4 f0 = *(const float4*)(src + c);
      float4 f1 = *(const float4*)(src + c + 4);
      store_bf8(dst + c, f0, f1);
    }
  }

  v8f acc[8] = {};   // 8 persistent 16x16 y-tiles per wave (fixed M-tile, 8 N-tiles)
  const int mt  = (wave & 3) * 16;
  const int nt0 = ((wave >> 2) * 2) * 16;     // GEMM1 N-tile pair base

  for (int c = 0; c < NCHUNK; ++c) {
    const int fc = c * KC;

    // --- stage pre-packed weight images: straight b128 memcpy ---
    {
      const uint4* s1 = (const uint4*)(w1t + (size_t)(e * NCHUNK + c) * W1_IMG_ELEMS);
      uint4* d1 = (uint4*)sW1;
      for (int idx = tid; idx < W1_IMG_ELEMS / 8; idx += 256) d1[idx] = s1[idx];
      const uint4* s2 = (const uint4*)(w2t + (size_t)(e * NCHUNK + c) * W2_IMG_ELEMS);
      uint4* d2 = (uint4*)sW2;
      #pragma unroll
      for (int idx = tid; idx < W2_IMG_ELEMS / 8; idx += 256) d2[idx] = s2[idx];
      if (tid < KC) sB1[tid] = b1[e * DF + fc + tid];
    }
    __syncthreads();

    // --- GEMM1: H[64xKC] = relu(X[64x256] @ W1chunk + b1chunk) -> sH (bf16) ---
    // One A load per k-step feeds both N-tiles (two chained accumulators).
    {
      v8f c0 = {}, c1 = {};
      #pragma unroll
      for (int kk = 0; kk < DM; kk += 32) {
        v16bf a  = load_a16x32(sX,  mt, SDM, kk, lane);
        v16bf b0 = load_b32x16(sW1, nt0,      SDM, kk, lane);
        v16bf b1v = load_b32x16(sW1, nt0 + 16, SDM, kk, lane);
        c0 = __builtin_amdgcn_wmma_f32_16x16x32_bf16(false, a, false, b0,
                                                     (short)0, c0, false, false);
        c1 = __builtin_amdgcn_wmma_f32_16x16x32_bf16(false, a, false, b1v,
                                                     (short)0, c1, false, false);
      }
      const int rbase = mt + ((lane >> 4) << 3);
      const int nc0   = nt0 + (lane & 15);
      const float bias0 = sB1[nc0];
      const float bias1 = sB1[nc0 + 16];
      #pragma unroll
      for (int r = 0; r < 8; ++r) {
        float v0 = c0[r] + bias0;
        float v1 = c1[r] + bias1;
        v0 = v0 > 0.f ? v0 : 0.f;
        v1 = v1 > 0.f ? v1 : 0.f;
        sH[(rbase + r) * SKC + nc0]      = (__bf16)v0;
        sH[(rbase + r) * SKC + nc0 + 16] = (__bf16)v1;
      }
    }
    __syncthreads();

    // --- GEMM2: acc += H[64xKC] @ W2chunk[KCx256] ---
    #pragma unroll
    for (int ks = 0; ks < KC; ks += 32) {
      v16bf a = load_a16x32(sH, mt, SKC, ks, lane);
      #pragma unroll
      for (int j = 0; j < 8; ++j) {
        const int nt = ((wave >> 2) * 8 + j) * 16;
        v16bf b = load_b32x16(sW2, nt, SKC, ks, lane);
        acc[j] = __builtin_amdgcn_wmma_f32_16x16x32_bf16(false, a, false, b,
                                                         (short)0, acc[j], false, false);
      }
    }
    __syncthreads();   // protect LDS before next chunk's staging
  }

  // --- epilogue: +b2, scatter by token id, mask tail rows ---
  const int rbase = mt + ((lane >> 4) << 3);
  #pragma unroll
  for (int j = 0; j < 8; ++j) {
    const int ncol   = ((wave >> 2) * 8 + j) * 16 + (lane & 15);
    const float bias = b2[e * DM + ncol];
    #pragma unroll
    for (int r = 0; r < 8; ++r) {
      const int row = rbase + r;
      if (tileBase + row < cnt) {
        out[(size_t)sTid[row] * DM + ncol] = acc[j][r] + bias;
      }
    }
  }
}

// ---------------- launch ----------------
extern "C" void kernel_launch(void* const* d_in, const int* in_sizes, int n_in,
                              void* d_out, int out_size, void* d_ws, size_t ws_size,
                              hipStream_t stream) {
  const float* x    = (const float*)d_in[0];
  const int*   bseq = (const int*)  d_in[1];
  const float* W1   = (const float*)d_in[2];
  const float* b1   = (const float*)d_in[3];
  const float* W2   = (const float*)d_in[4];
  const float* b2   = (const float*)d_in[5];
  float* out = (float*)d_out;

  int*    counts = (int*)((char*)d_ws + WS_COUNTS);
  int*    lists  = (int*)((char*)d_ws + WS_LISTS);
  __bf16* w1t    = (__bf16*)((char*)d_ws + WS_W1T);
  __bf16* w2t    = (__bf16*)((char*)d_ws + WS_W2T);

  // Pass 0: zero output (covers expert-0 tokens) and counters
  const int n4 = NTOK * DM / 4;
  bpff_zero_kernel<<<(n4 + 255) / 256, 256, 0, stream>>>((float4*)out, n4, counts);

  // Pass 1: build per-expert token lists
  bpff_build_lists<<<NTOK / 256, 256, 0, stream>>>(bseq, counts, lists);

  // Pass 1.5: pack weights into bf16 transposed LDS-image layout (once per call)
  bpff_pack_weights<<<dim3(NCHUNK, NB), 256, 0, stream>>>(W1, W2, w1t, w2t);

  // Pass 2: gathered grouped-GEMM FFN (bf16 WMMA, f32 accumulate)
  constexpr int SMEM_BYTES =
      512 + (TM * SDM + KC * SDM + DM * SKC + TM * SKC) * (int)sizeof(__bf16);
  dim3 grid(NTOK / TM, NB);
  bpff_ffn_kernel<<<grid, 256, SMEM_BYTES, stream>>>(x, w1t, b1, w2t, b2,
                                                     counts, lists, out);
}
